// TrigoAttention_37520834298118
// MI455X (gfx1250) — matrix operand order repaired and checked by reading, not compiled, not run on hardware
//
#include <hip/hip_runtime.h>

// ---------------------------------------------------------------------------
// TrigoAttention for MI455X (gfx1250, wave32)
//   D_MODEL=256, HEADS=8, D_K=32, bs=4, seq=512, N = bs*seq = 2048
//
// Cost model: 4 ripple layers = ~537M sin+fma groups  -> VALU/trans bound.
// Attention = ~1 GFLOP of GEMM -> v_wmma_f32_16x16x32_f16 (f32 accumulate).
// All data fits in L2 (192MB); HBM irrelevant.
// ---------------------------------------------------------------------------

typedef _Float16 v16h __attribute__((ext_vector_type(16)));
typedef _Float16 v8h  __attribute__((ext_vector_type(8)));
typedef float    v8f  __attribute__((ext_vector_type(8)));

#define NTOT   2048      // bs*seq
#define DM     256
#define SEQ    512
#define NBH    32        // bs*heads
#define DK     32
#define KI     32        // i-chunk staged in LDS

#define INV2PI    0.15915494309189535f   // fold into w1/b so v_sin gets revolutions
#define QK_SCALE  0.17677669529663687f   // 1/sqrt(32), folded into Q at f16 store
#define LOG2E     1.4426950408889634f

// ---------------------------------------------------------------------------
// Ripple linear: y[n,o] = sum_i w0[o,i]*sin(w1[o,i]*x[n,i]+b[o,i]) + b[o,-1]
// Block tile 64n x 64o, 256 threads, 4x4 register tile per thread.
// mode 0: f32 out [n, o]                      (final projection -> d_out)
// mode 1: f16 Qh [b,h,s,d], scaled by 1/sqrt(dk)
// mode 2: f16 Kh [b,h,s,d]
// mode 3: f16 Vt [b,h,d,s]  (pre-transposed so attention B-loads are linear)
// ---------------------------------------------------------------------------
__global__ __launch_bounds__(256) void ripple_proj(
    const float* __restrict__ x,     // [NTOT, 256]
    const float* __restrict__ w,     // [256, 256, 2]  (w0 = [...,0], w1 = [...,1])
    const float* __restrict__ bias,  // [256, 257]
    float*    __restrict__ out_f32,
    _Float16* __restrict__ out_f16,
    int mode)
{
    __shared__ float xs  [64][KI + 1];
    __shared__ float w0s [64][KI];
    __shared__ float w1ls[64][KI];
    __shared__ float bls [64][KI];

    const int t  = threadIdx.x;
    const int n0 = blockIdx.x * 64;
    const int o0 = blockIdx.y * 64;
    const int tn = t & 15;   // 4 consecutive n per thread
    const int to = t >> 4;   // 4 consecutive o per thread

    float acc[4][4];
#pragma unroll
    for (int a = 0; a < 4; ++a)
#pragma unroll
        for (int r = 0; r < 4; ++r) acc[a][r] = 0.0f;

    for (int i0 = 0; i0 < DM; i0 += KI) {
        __syncthreads();
        // stage x tile (64 x KI) and weight tiles (64 x KI x 3); scale by 1/2pi here
        for (int l = t; l < 64 * KI; l += 256) {
            int nl = l >> 5, il = l & 31;
            xs[nl][il] = x[(size_t)(n0 + nl) * DM + i0 + il];
        }
        for (int l = t; l < 64 * KI; l += 256) {
            int ol = l >> 5, il = l & 31;
            int gi = (o0 + ol) * DM + i0 + il;
            w0s [ol][il] = w[(size_t)gi * 2 + 0];
            w1ls[ol][il] = w[(size_t)gi * 2 + 1] * INV2PI;
            bls [ol][il] = bias[(size_t)(o0 + ol) * (DM + 1) + i0 + il] * INV2PI;
        }
        __syncthreads();

        // overlap next chunk's cacheline fill with the trans-bound inner loop
        // (lowers to global_prefetch_b8 on gfx1250)
        if (i0 + KI < DM) {
            const int nl = t & 63;
            __builtin_prefetch(&x[(size_t)(n0 + nl) * DM + i0 + KI], 0, 3);
            __builtin_prefetch(&w[((size_t)(o0 + nl) * DM + i0 + KI) * 2], 0, 3);
            __builtin_prefetch(&bias[(size_t)(o0 + nl) * (DM + 1) + i0 + KI], 0, 3);
        }

        for (int ii = 0; ii < KI; ++ii) {
            float xr[4];
#pragma unroll
            for (int r = 0; r < 4; ++r) xr[r] = xs[tn * 4 + r][ii];
#pragma unroll
            for (int oo = 0; oo < 4; ++oo) {
                const int ol = to * 4 + oo;
                const float w1v = w1ls[ol][ii];
                const float bv  = bls [ol][ii];
                const float w0v = w0s [ol][ii];
#pragma unroll
                for (int r = 0; r < 4; ++r) {
                    // v_sin_f32 input is in revolutions; 1/2pi already folded in
                    float s = __builtin_amdgcn_sinf(fmaf(w1v, xr[r], bv));
                    acc[oo][r] = fmaf(w0v, s, acc[oo][r]);
                }
            }
        }
    }

#pragma unroll
    for (int oo = 0; oo < 4; ++oo) {
        const int o  = o0 + to * 4 + oo;
        const float bf = bias[(size_t)o * (DM + 1) + DM];
#pragma unroll
        for (int r = 0; r < 4; ++r) {
            const int n = n0 + tn * 4 + r;
            const float y = acc[oo][r] + bf;
            if (mode == 0) {
                out_f32[(size_t)n * DM + o] = y;
            } else {
                const int b = n >> 9, s = n & 511;
                const int h = o >> 5, d = o & 31;
                if (mode == 1)
                    out_f16[(((size_t)(b * 8 + h) * SEQ) + s) * DK + d] =
                        (_Float16)(y * QK_SCALE);
                else if (mode == 2)
                    out_f16[(((size_t)(b * 8 + h) * SEQ) + s) * DK + d] = (_Float16)y;
                else // mode 3: transposed V  [bh][d][s]
                    out_f16[(((size_t)(b * 8 + h) * DK) + d) * SEQ + s] = (_Float16)y;
            }
        }
    }
}

// ---------------------------------------------------------------------------
// Attention: one workgroup = 16 query rows of one (b,h); 8 waves, 256 threads.
//   phase 1: S[16,512] = Q Kt  via v_wmma_f32_16x16x32_f16 (4 tiles / wave,
//            fixed trip count so the compiler fully unrolls -> EXEC stays all-1s)
//   phase 2: row softmax in f32 (exp2 * log2e), P stored f16 in LDS
//   phase 3: O[16,32] = P V    via WMMA; 2 d-halves x 4 kv-quarters over waves,
//            cross-wave LDS reduction + 1/rowsum scaling
// ---------------------------------------------------------------------------
__global__ __launch_bounds__(256) void attn_wmma(
    const _Float16* __restrict__ Qh,   // [NBH][SEQ][DK]  (pre-scaled)
    const _Float16* __restrict__ Kh,   // [NBH][SEQ][DK]
    const _Float16* __restrict__ Vt,   // [NBH][DK][SEQ]
    float* __restrict__ concat)        // [NTOT][DM]
{
    __shared__ float    sS[16][SEQ];        // 32 KB
    __shared__ _Float16 sP[16][SEQ];        // 16 KB
    __shared__ float    redmax[16][16];
    __shared__ float    redsum[16][16];
    __shared__ float    sO[4][16][DK];      //  8 KB

    const int t    = threadIdx.x;
    const int lane = t & 31;
    const int wave = t >> 5;
    const int bh   = blockIdx.x >> 5;   // 0..31
    const int qt   = blockIdx.x & 31;   // 16-row query tile
    const int qr0  = qt * 16;
    const int lrow = lane & 15;
    const int lhi  = lane >> 4;         // half-wave select per ISA A/B layouts

    // ---- A operand: Q rows (16x32 f16). lane<16 holds K {0..15},
    //      lane>=16 holds K {16..31}, row M = lane%16. One 32B load. ----
    v16h qa;
    {
        const _Float16* Qp = Qh + ((size_t)bh * SEQ + qr0 + lrow) * DK;
        qa = *(const v16h*)(Qp + lhi * 16);
    }

    // ---- phase 1: S = Q Kt  (each wave: kv tiles wave, wave+8, wave+16, wave+24)
    const _Float16* Kbase = Kh + (size_t)bh * SEQ * DK;
#pragma unroll
    for (int tt = 0; tt < 4; ++tt) {
        const int jt = tt * 8 + wave;   // < 32 for every wave: trip count is static
        // B operand (32x16): lane holds column N=kv (=jt*16+lrow),
        // K-range lhi*16..+15 -> contiguous 32B in Kh row-major [s][d].
        const int kv = jt * 16 + lrow;
        v16h kb = *(const v16h*)(Kbase + (size_t)kv * DK + lhi * 16);
        v8f c = {};
        c = __builtin_amdgcn_wmma_f32_16x16x32_f16(false, qa, false, kb,
                                                   (short)0, c, false, false);
        // C/D layout: VGPR r -> M = r + lhi*8, N = lrow
#pragma unroll
        for (int r = 0; r < 8; ++r)
            sS[r + lhi * 8][jt * 16 + lrow] = c[r];
    }
    __syncthreads();

    // ---- phase 2: softmax rows (16 threads per row, 32 cols each) ----
    const int srow = t >> 4;
    const int sseg = t & 15;
    {
        float m = -3.0e38f;
        for (int c0 = 0; c0 < 32; ++c0)
            m = fmaxf(m, sS[srow][sseg * 32 + c0]);
        redmax[srow][sseg] = m;
    }
    __syncthreads();
    {
        float rowm = -3.0e38f;
#pragma unroll
        for (int k = 0; k < 16; ++k) rowm = fmaxf(rowm, redmax[srow][k]);
        float sum = 0.0f;
        for (int c0 = 0; c0 < 32; ++c0) {
            float e = __builtin_amdgcn_exp2f((sS[srow][sseg * 32 + c0] - rowm) * LOG2E);
            sP[srow][sseg * 32 + c0] = (_Float16)e;
            sum += e;
        }
        redsum[srow][sseg] = sum;
    }
    __syncthreads();

    // ---- phase 3: O = P V (unnormalized), waves split 2 d-halves x 4 kv-quarters
    {
        const int ntile = wave & 1;   // which 16 of the 32 d-columns
        const int kq    = wave >> 1;  // which 128-kv quarter
        const _Float16* Vbase =
            Vt + (size_t)bh * DK * SEQ + (size_t)(ntile * 16 + lrow) * SEQ;
        v8f c = {};
#pragma unroll
        for (int s = 0; s < 4; ++s) {
            const int k0 = kq * 128 + s * 32;
            v16h pa = *(const v16h*)&sP[lrow][k0 + lhi * 16];
            v16h vb = *(const v16h*)(Vbase + k0 + lhi * 16);
            c = __builtin_amdgcn_wmma_f32_16x16x32_f16(false, pa, false, vb,
                                                       (short)0, c, false, false);
        }
#pragma unroll
        for (int r = 0; r < 8; ++r)
            sO[kq][r + lhi * 8][ntile * 16 + lrow] = c[r];
    }
    __syncthreads();

    // ---- reduce quarters, normalize, write concat[b][s][h*32+d] ----
    {
        const int d  = t & 31;
        const int r0 = t >> 5;        // 0..7, handles rows r0 and r0+8
        const int b = bh >> 3, h = bh & 7;
#pragma unroll
        for (int rr = 0; rr < 2; ++rr) {
            const int row = r0 + rr * 8;
            float rs = 0.0f;
#pragma unroll
            for (int k = 0; k < 16; ++k) rs += redsum[row][k];
            const float o = (sO[0][row][d] + sO[1][row][d] +
                             sO[2][row][d] + sO[3][row][d]) / rs;
            concat[((size_t)(b * SEQ + qr0 + row)) * DM + h * DK + d] = o;
        }
    }
}

// ---------------------------------------------------------------------------
extern "C" void kernel_launch(void* const* d_in, const int* in_sizes, int n_in,
                              void* d_out, int out_size, void* d_ws, size_t ws_size,
                              hipStream_t stream) {
    (void)in_sizes; (void)n_in; (void)out_size; (void)ws_size;
    const float* q  = (const float*)d_in[0];
    const float* k  = (const float*)d_in[1];
    const float* v  = (const float*)d_in[2];
    const float* wq = (const float*)d_in[3];
    const float* bq = (const float*)d_in[4];
    const float* wk = (const float*)d_in[5];
    const float* bk = (const float*)d_in[6];
    const float* wv = (const float*)d_in[7];
    const float* bv = (const float*)d_in[8];
    const float* wo = (const float*)d_in[9];
    const float* bo = (const float*)d_in[10];

    // workspace layout (~5 MB)
    char* ws = (char*)d_ws;
    size_t off = 0;
    _Float16* Qh = (_Float16*)(ws + off); off += (size_t)NBH * SEQ * DK * 2;
    _Float16* Kh = (_Float16*)(ws + off); off += (size_t)NBH * SEQ * DK * 2;
    _Float16* Vt = (_Float16*)(ws + off); off += (size_t)NBH * DK * SEQ * 2;
    float* concat = (float*)(ws + off);   off += (size_t)NTOT * DM * 4;

    dim3 pg(NTOT / 64, DM / 64);   // (32, 4)
    ripple_proj<<<pg, 256, 0, stream>>>(q, wq, bq, nullptr, Qh, 1);
    ripple_proj<<<pg, 256, 0, stream>>>(k, wk, bk, nullptr, Kh, 2);
    ripple_proj<<<pg, 256, 0, stream>>>(v, wv, bv, nullptr, Vt, 3);

    attn_wmma<<<NBH * (SEQ / 16), 256, 0, stream>>>(Qh, Kh, Vt, concat);

    ripple_proj<<<pg, 256, 0, stream>>>(concat, wo, bo, (float*)d_out, nullptr, 0);
}
